// _Reformer_4896262717676
// MI455X (gfx1250) — matrix operand order repaired and checked by reading, not compiled, not run on hardware
//
#include <hip/hip_runtime.h>
#include <hip/hip_bf16.h>

typedef __attribute__((ext_vector_type(16))) _Float16 v16h;
typedef __attribute__((ext_vector_type(4)))  _Float16 v4h;
typedef __attribute__((ext_vector_type(8)))  float    v8f;

// ---------------- constants ----------------
#define BSZ 4
#define LSEQ 4096
#define LENC 3072
#define PREDL 1024
#define DMODEL 512
#define NHEADS 8
#define DHEAD 64
#define DFF 2048
#define BH (BSZ*NHEADS)          // 32
#define NHASH 4
#define NBUCK 64                 // buckets per hash round
#define NSORT (NHASH*LSEQ)       // 16384 items per bh row
#define NCH 256                  // chunks per bh row (NSORT/64)
#define SELF_MASK (-5e4f)

// ---------------- helpers ----------------
__device__ __forceinline__ v8f zero8() {
  v8f v;
#pragma unroll
  for (int i = 0; i < 8; ++i) v[i] = 0.0f;
  return v;
}

__device__ __forceinline__ float gelu_tanh(float x) {
  const float c = 0.7978845608028654f;
  float t = tanhf(c * (x + 0.044715f * x * x * x));
  return 0.5f * x * (1.0f + t);
}

// A-operand LDS swizzle: identity map (8-blocks keep their place), so the
// A-fragment is the contiguous run k = hi*16 + e, same order as B (not the ISA table order).
// Involution: write with it, read linearly.
__device__ __forceinline__ int swz32(int c) {
  int blk = (c >> 3) & 3;
  int pb = blk;                                // 0,1,2,3
  return (c & ~31) | (pb << 3) | (c & 7);
}

// Unified fragment load: one aligned 32-byte LDS read per lane.
// A role: S stored swizzled (swz32 on write), row = M.
// B role: S stored transposed [N][K] natural order, row = N.
__device__ __forceinline__ v16h frag_ld(const _Float16* S, int ld, int rBase, int k0, int lane) {
  int r = rBase + (lane & 15);
  int hi = lane >> 4;
  return *((const v16h*)(S + r * ld + k0 + (hi << 4)));
}

__device__ __forceinline__ v4h cvt4(float4 f) {
  v4h h;
  h[0] = (_Float16)f.x; h[1] = (_Float16)f.y;
  h[2] = (_Float16)f.z; h[3] = (_Float16)f.w;
  return h;
}

// ---------------- embedding ----------------
__global__ __launch_bounds__(256) void embed_kernel(
    const float* __restrict__ xe, const float* __restrict__ xd,
    const float* __restrict__ me, const float* __restrict__ md,
    const float* __restrict__ conv, const float* __restrict__ tw,
    float* __restrict__ h) {
  int g = blockIdx.x * 256 + threadIdx.x;      // 16384*512 threads
  int d = g & 511;
  int row = g >> 9;
  int b = row >> 12;
  int l = row & 4095;
  float acc = 0.0f;
#pragma unroll
  for (int w = 0; w < 3; ++w) {
    int lp = (l - 1 + w + LSEQ) & (LSEQ - 1);
#pragma unroll
    for (int c = 0; c < 7; ++c) {
      float xv = (lp < LENC) ? xe[(b * LENC + lp) * 7 + c]
                             : xd[(b * 1536 + (lp - LENC + 512)) * 7 + c];
      acc += xv * conv[(w * 7 + c) * DMODEL + d];
    }
  }
  // sinusoidal positional embedding
  float dv = expf(-(float)(d & ~1) * (9.2103403719761836f / 512.0f));
  float ang = (float)l * dv;
  acc += (d & 1) ? cosf(ang) : sinf(ang);
  // temporal marks
#pragma unroll
  for (int k = 0; k < 4; ++k) {
    float mv = (l < LENC) ? me[(b * LENC + l) * 4 + k]
                          : md[(b * 1536 + (l - LENC + 512)) * 4 + k];
    acc += mv * tw[k * DMODEL + d];
  }
  h[(size_t)row * DMODEL + d] = acc;
}

// ---------------- WMMA GEMM: C = act(A @ W + bias) ----------------
// A [M,K] f32, W [K,N] f32, C [M,N] f32. Block tile 128x64, BK=64, 8 waves.
__global__ __launch_bounds__(256) void gemm_wmma_kernel(
    const float* __restrict__ A, const float* __restrict__ W,
    const float* __restrict__ bias, float* __restrict__ C,
    int M, int N, int K, int act) {
  __shared__ _Float16 As[128 * 64];    // swizzled A tile (16KB)
  __shared__ _Float16 WsT[64 * 64];    // transposed W tile [n][k] (8KB)
  int tid = threadIdx.x, lane = tid & 31, wid = tid >> 5;
  int bm = blockIdx.y * 128, bn = blockIdx.x * 64;
  int wm = (wid & 3) * 32, wn = (wid >> 2) * 32;
  v8f acc00 = zero8(), acc01 = zero8(), acc10 = zero8(), acc11 = zero8();

  for (int k0 = 0; k0 < K; k0 += 64) {
    // A tile: float4 global reads, packed swizzled v4h LDS stores
    for (int slot = tid; slot < 128 * 16; slot += 256) {
      int r = slot >> 4;
      int c4 = (slot & 15) << 2;
      float4 f = *(const float4*)(A + (size_t)(bm + r) * K + k0 + c4);
      *(v4h*)(As + r * 64 + swz32(c4)) = cvt4(f);
    }
    // W tile: float4 global reads, transposed scalar LDS stores
    for (int slot = tid; slot < 64 * 16; slot += 256) {
      int r = slot >> 4;                 // k within chunk
      int c4 = (slot & 15) << 2;         // n
      float4 f = *(const float4*)(W + (size_t)(k0 + r) * N + bn + c4);
      WsT[(c4 + 0) * 64 + r] = (_Float16)f.x;
      WsT[(c4 + 1) * 64 + r] = (_Float16)f.y;
      WsT[(c4 + 2) * 64 + r] = (_Float16)f.z;
      WsT[(c4 + 3) * 64 + r] = (_Float16)f.w;
    }
    __syncthreads();
#pragma unroll
    for (int ks = 0; ks < 64; ks += 32) {
      v16h a0 = frag_ld(As, 64, wm + 0, ks, lane);
      v16h a1 = frag_ld(As, 64, wm + 16, ks, lane);
      v16h b0 = frag_ld(WsT, 64, wn + 0, ks, lane);
      v16h b1 = frag_ld(WsT, 64, wn + 16, ks, lane);
      acc00 = __builtin_amdgcn_wmma_f32_16x16x32_f16(false, a0, false, b0, (short)0, acc00, false, false);
      acc01 = __builtin_amdgcn_wmma_f32_16x16x32_f16(false, a0, false, b1, (short)0, acc01, false, false);
      acc10 = __builtin_amdgcn_wmma_f32_16x16x32_f16(false, a1, false, b0, (short)0, acc10, false, false);
      acc11 = __builtin_amdgcn_wmma_f32_16x16x32_f16(false, a1, false, b1, (short)0, acc11, false, false);
    }
    __syncthreads();
  }

#pragma unroll
  for (int tm = 0; tm < 2; ++tm) {
#pragma unroll
    for (int tn = 0; tn < 2; ++tn) {
      v8f acc = (tm == 0) ? ((tn == 0) ? acc00 : acc01) : ((tn == 0) ? acc10 : acc11);
#pragma unroll
      for (int r = 0; r < 8; ++r) {
        int m = bm + wm + tm * 16 + ((lane >> 4) << 3) + r;
        int n = bn + wn + tn * 16 + (lane & 15);
        float v = acc[r];
        if (bias) v += bias[n];
        if (act) v = gelu_tanh(v);
        C[(size_t)m * N + n] = v;
      }
    }
  }
}

// ---------------- LSH hashing ----------------
__global__ __launch_bounds__(256) void hash_kernel(
    const float* __restrict__ qk, const float* __restrict__ rot,
    int* __restrict__ bktid) {
  int g = blockIdx.x * 256 + threadIdx.x;      // BH*L = 131072 threads
  int bh = g >> 12;
  int l = g & 4095;
  int b = bh >> 3, hd = bh & 7;
  const float* q = qk + ((size_t)(b * LSEQ + l)) * DMODEL + hd * DHEAD;
  float qr[DHEAD];
#pragma unroll 8
  for (int e = 0; e < DHEAD; ++e) qr[e] = q[e];
  for (int r = 0; r < NHASH; ++r) {
    float dots[32];
    for (int i = 0; i < 32; ++i) {
      float d = 0.0f;
#pragma unroll 8
      for (int e = 0; e < DHEAD; ++e) d += qr[e] * rot[(e * NHASH + r) * 32 + i];
      dots[i] = d;
    }
    float best = -__builtin_inff();
    int bi = 0;
    for (int i = 0; i < 64; ++i) {            // first-max tie-break like jnp.argmax
      float v = (i < 32) ? dots[i] : -dots[i - 32];
      if (v > best) { best = v; bi = i; }
    }
    bktid[bh * NSORT + r * LSEQ + l] = bi + r * NBUCK;
  }
}

// ---------------- counting-sort: histogram + exclusive scan ----------------
__global__ __launch_bounds__(256) void hist_kernel(
    const int* __restrict__ bktid, int* __restrict__ offs) {
  int bh = blockIdx.x;
  __shared__ int hist[256];
  hist[threadIdx.x] = 0;
  __syncthreads();
  for (int i = threadIdx.x; i < NSORT; i += 256)
    atomicAdd(&hist[bktid[bh * NSORT + i]], 1);
  __syncthreads();
  if (threadIdx.x == 0) {
    int acc = 0;
    for (int k = 0; k < 256; ++k) { offs[bh * 257 + k] = acc; acc += hist[k]; }
    offs[bh * 257 + 256] = acc;
  }
}

// stable scatter: one wave per (bh, bucket); builds permutation + inverse
__global__ __launch_bounds__(32) void sort_kernel(
    const int* __restrict__ bktid, const int* __restrict__ offs,
    int* __restrict__ sorted, int* __restrict__ inv) {
  int bh = blockIdx.x >> 8;
  int bin = blockIdx.x & 255;
  int lane = threadIdx.x;
  int base = offs[bh * 257 + bin];
  const int* bp = bktid + bh * NSORT;
  for (int i0 = 0; i0 < NSORT; i0 += 32) {
    int i = i0 + lane;
    bool p = (bp[i] == bin);
    unsigned mask = __builtin_amdgcn_ballot_w32(p);
    int rank = __popc(mask & ((1u << lane) - 1u));
    if (p) {
      int s = base + rank;
      sorted[bh * NSORT + s] = i;
      inv[bh * NSORT + i] = s;
    }
    base += __popc(mask);
  }
}

// ---------------- chunked LSH attention (WMMA) ----------------
__global__ __launch_bounds__(256) void lsh_attn_kernel(
    const float* __restrict__ qk, const float* __restrict__ vv,
    const int* __restrict__ sorted, float* __restrict__ so,
    float* __restrict__ slog) {
  __shared__ __align__(32) char smem[56 * 1024];
  __shared__ int qgid[64];
  __shared__ int kgid[128];
  __shared__ float knorm[128];
  __shared__ float rowlse[64];
  _Float16* Qs  = (_Float16*)smem;             // [64][64] swizzled  (phase 1)
  _Float16* Ks  = (_Float16*)(smem + 8192);    // [128][64] natural  (phase 1, B role)
  float*    Ds  = (float*)(smem + 24576);      // [64][128]          (phase 1/2)
  _Float16* VsT = (_Float16*)smem;             // [64][128] = [e][j] (phase 2, B role)
  _Float16* Ps  = (_Float16*)(smem + 24576);   // [64][128] swizzled (phase 2, A role)

  int tid = threadIdx.x, lane = tid & 31, wid = tid >> 5;
  int bh = blockIdx.x >> 8;
  int c = blockIdx.x & 255;
  int b = bh >> 3, hd = bh & 7;
  int s0 = c * 64;
  int cprev = (c + NCH - 1) & (NCH - 1);
  const int* srt = sorted + bh * NSORT;

  if (tid < 64) qgid[tid] = srt[s0 + tid];
  if (tid < 128) {
    int jc = (tid < 64) ? c : cprev;           // [chunk, look-one-back]
    kgid[tid] = srt[jc * 64 + (tid & 63)];
  }
  __syncthreads();

  // Q tile (f16, swizzled for A role); float4 global reads
  for (int slot = tid; slot < 64 * 16; slot += 256) {
    int row = slot >> 4;
    int e4 = (slot & 15) << 2;
    int l = qgid[row] & (LSEQ - 1);
    float4 f = *(const float4*)(qk + ((size_t)(b * LSEQ + l)) * DMODEL + hd * DHEAD + e4);
    *(v4h*)(Qs + row * 64 + swz32(e4)) = cvt4(f);
  }
  // key norms (f32, float4 reads)
  if (tid < 128) {
    int l = kgid[tid] & (LSEQ - 1);
    const float* kr = qk + ((size_t)(b * LSEQ + l)) * DMODEL + hd * DHEAD;
    float s = 0.0f;
#pragma unroll
    for (int e4 = 0; e4 < 64; e4 += 4) {
      float4 f = *(const float4*)(kr + e4);
      s += f.x * f.x + f.y * f.y + f.z * f.z + f.w * f.w;
    }
    knorm[tid] = fmaxf(sqrtf(s), 1e-12f);
  }
  __syncthreads();
  // normalized shared-QK keys (f16, natural [j][k] layout for B role)
  for (int slot = tid; slot < 128 * 16; slot += 256) {
    int j = slot >> 4;
    int e4 = (slot & 15) << 2;
    int l = kgid[j] & (LSEQ - 1);
    float rn = 1.0f / knorm[j];
    float4 f = *(const float4*)(qk + ((size_t)(b * LSEQ + l)) * DMODEL + hd * DHEAD + e4);
    f.x *= rn; f.y *= rn; f.z *= rn; f.w *= rn;
    *(v4h*)(Ks + j * 64 + e4) = cvt4(f);
  }
  __syncthreads();

  // dots 64x128 = 32 tiles of 16x16; 8 waves * 4 tiles; K=64 (2 wmma steps)
  for (int q = 0; q < 4; ++q) {
    int t = wid * 4 + q;
    int mt = t >> 3, nt = t & 7;
    v8f acc = zero8();
#pragma unroll
    for (int k0 = 0; k0 < 64; k0 += 32) {
      v16h a = frag_ld(Qs, 64, mt * 16, k0, lane);
      v16h bf = frag_ld(Ks, 64, nt * 16, k0, lane);
      acc = __builtin_amdgcn_wmma_f32_16x16x32_f16(false, a, false, bf, (short)0, acc, false, false);
    }
#pragma unroll
    for (int r = 0; r < 8; ++r) {
      int m = mt * 16 + ((lane >> 4) << 3) + r;
      int n = nt * 16 + (lane & 15);
      Ds[m * 128 + n] = acc[r] * 0.125f;       // * D_HEAD^-0.5
    }
  }
  __syncthreads();

  // self-position mask + logsumexp per row
  if (tid < 64) {
    int qp = qgid[tid] & (LSEQ - 1);
    float mx = -__builtin_inff();
    for (int j = 0; j < 128; ++j) {
      float d = Ds[tid * 128 + j];
      if ((kgid[j] & (LSEQ - 1)) == qp) d = SELF_MASK;
      Ds[tid * 128 + j] = d;
      mx = fmaxf(mx, d);
    }
    float sum = 0.0f;
    for (int j = 0; j < 128; ++j) sum += expf(Ds[tid * 128 + j] - mx);
    float lse = mx + logf(sum);
    rowlse[tid] = lse;
    slog[bh * NSORT + s0 + tid] = lse;
  }
  __syncthreads();

  // stage probs into registers (Ds still live); load V transposed over Qs/Ks
  float pr[32];
  {
    int base = tid * 32;
#pragma unroll
    for (int u = 0; u < 32; ++u) {
      int idx = base + u;
      pr[u] = expf(Ds[idx] - rowlse[idx >> 7]);
    }
  }
  for (int slot = tid; slot < 128 * 16; slot += 256) {
    int j = slot >> 4;
    int e4 = (slot & 15) << 2;
    int l = kgid[j] & (LSEQ - 1);
    float4 f = *(const float4*)(vv + ((size_t)(b * LSEQ + l)) * DMODEL + hd * DHEAD + e4);
    VsT[(e4 + 0) * 128 + j] = (_Float16)f.x;
    VsT[(e4 + 1) * 128 + j] = (_Float16)f.y;
    VsT[(e4 + 2) * 128 + j] = (_Float16)f.z;
    VsT[(e4 + 3) * 128 + j] = (_Float16)f.w;
  }
  __syncthreads();
  {
    int base = tid * 32;
#pragma unroll
    for (int u = 0; u < 32; ++u) {
      int idx = base + u;
      int row = idx >> 7, col = idx & 127;
      Ps[row * 128 + swz32(col)] = (_Float16)pr[u];
    }
  }
  __syncthreads();

  // out 64x64 = 16 tiles; 8 waves * 2 tiles; K=128 (4 wmma steps)
  for (int q = 0; q < 2; ++q) {
    int t = wid * 2 + q;
    int mt = t >> 2, nt = t & 3;
    v8f acc = zero8();
#pragma unroll
    for (int k0 = 0; k0 < 128; k0 += 32) {
      v16h a = frag_ld(Ps, 128, mt * 16, k0, lane);
      v16h bf = frag_ld(VsT, 128, nt * 16, k0, lane);
      acc = __builtin_amdgcn_wmma_f32_16x16x32_f16(false, a, false, bf, (short)0, acc, false, false);
    }
#pragma unroll
    for (int r = 0; r < 8; ++r) {
      int m = mt * 16 + ((lane >> 4) << 3) + r;
      int n = nt * 16 + (lane & 15);
      so[((size_t)bh * NSORT + s0 + m) * DHEAD + n] = acc[r];
    }
  }
}

// ---------------- combine hash rounds (softmax over logits) ----------------
__global__ __launch_bounds__(256) void combine_kernel(
    const float* __restrict__ so, const float* __restrict__ slog,
    const int* __restrict__ inv, float* __restrict__ out) {
  int pair = blockIdx.x * 4 + (threadIdx.x >> 6);   // (bh, l) pair
  int e = threadIdx.x & 63;
  int bh = pair >> 12;
  int l = pair & 4095;
  int b = bh >> 3, hd = bh & 7;
  int s[NHASH];
  float lg[NHASH];
  float mx = -__builtin_inff();
#pragma unroll
  for (int r = 0; r < NHASH; ++r) {
    s[r] = inv[bh * NSORT + r * LSEQ + l];
    lg[r] = slog[bh * NSORT + s[r]];
    mx = fmaxf(mx, lg[r]);
  }
  float den = 0.0f, w[NHASH];
#pragma unroll
  for (int r = 0; r < NHASH; ++r) { w[r] = expf(lg[r] - mx); den += w[r]; }
  float o = 0.0f;
#pragma unroll
  for (int r = 0; r < NHASH; ++r)
    o += (w[r] / den) * so[((size_t)bh * NSORT + s[r]) * DHEAD + e];
  out[((size_t)(b * LSEQ + l)) * DMODEL + hd * DHEAD + e] = o;
}

// ---------------- residual + LayerNorm (in place on h) ----------------
__global__ __launch_bounds__(256) void ln_kernel(
    float* __restrict__ h, const float* __restrict__ y,
    const float* __restrict__ g, const float* __restrict__ b) {
  int row = blockIdx.x;
  int tid = threadIdx.x;
  __shared__ float r1[256], r2[256];
  size_t base = (size_t)row * DMODEL;
  float v0 = h[base + tid] + (y ? y[base + tid] : 0.0f);
  float v1 = h[base + 256 + tid] + (y ? y[base + 256 + tid] : 0.0f);
  r1[tid] = v0 + v1;
  r2[tid] = v0 * v0 + v1 * v1;
  __syncthreads();
  for (int o = 128; o > 0; o >>= 1) {
    if (tid < o) { r1[tid] += r1[tid + o]; r2[tid] += r2[tid + o]; }
    __syncthreads();
  }
  float mu = r1[0] * (1.0f / DMODEL);
  float var = r2[0] * (1.0f / DMODEL) - mu * mu;
  float is = rsqrtf(var + 1e-5f);
  h[base + tid]       = (v0 - mu) * is * g[tid] + b[tid];
  h[base + 256 + tid] = (v1 - mu) * is * g[256 + tid] + b[256 + tid];
}

// ---------------- final projection (last PRED_LEN rows) ----------------
__global__ __launch_bounds__(256) void proj_kernel(
    const float* __restrict__ h, const float* __restrict__ W,
    const float* __restrict__ bias, float* __restrict__ out) {
  int g = blockIdx.x * 256 + threadIdx.x;      // 4*1024*7
  if (g >= BSZ * PREDL * 7) return;
  int c = g % 7;
  int t = g / 7;
  int b = t >> 10;
  int lp = t & 1023;
  const float* hr = h + ((size_t)(b * LSEQ + LENC + lp)) * DMODEL;
  float acc = bias[c];
  for (int d = 0; d < DMODEL; ++d) acc += hr[d] * W[d * 7 + c];
  out[g] = acc;
}

// ---------------- host orchestration ----------------
extern "C" void kernel_launch(void* const* d_in, const int* in_sizes, int n_in,
                              void* d_out, int out_size, void* d_ws, size_t ws_size,
                              hipStream_t stream) {
  (void)in_sizes; (void)n_in; (void)out_size; (void)ws_size;
  // setup_inputs() dict insertion order: x_enc, x_mark_enc, x_dec, x_mark_dec,
  // then params (insertion order): token_conv, temporal_W, proj_W, proj_b,
  // ln_f_g, ln_f_b, rotations, then 2 layers each:
  // Wqk, Wv, Wo, bo, ln1_g, ln1_b, ln2_g, ln2_b, W1, b1, W2, b2
  const float* x_enc      = (const float*)d_in[0];
  const float* x_mark_enc = (const float*)d_in[1];
  const float* x_dec      = (const float*)d_in[2];
  const float* x_mark_dec = (const float*)d_in[3];
  const float* token_conv = (const float*)d_in[4];
  const float* temporal_W = (const float*)d_in[5];
  const float* proj_W     = (const float*)d_in[6];
  const float* proj_b     = (const float*)d_in[7];
  const float* ln_f_g     = (const float*)d_in[8];
  const float* ln_f_b     = (const float*)d_in[9];
  const float* rotations  = (const float*)d_in[10];
  auto LW = [&](int lp, int j) { return (const float*)d_in[11 + lp * 12 + j]; };

  // workspace layout (floats)
  float* ws  = (float*)d_ws;
  const size_t NH = (size_t)BSZ * LSEQ * DMODEL;        // 8388608
  float* h   = ws;
  float* ta  = h + NH;
  float* tqk = ta + NH;
  float* tv  = tqk + NH;
  float* big = tv + NH;                                  // 33554432 (so / ffn hidden)
  float* slg = big + (size_t)BSZ * LSEQ * DFF;           // 524288
  int*   bkt = (int*)(slg + (size_t)BH * NSORT);
  int*   srt = bkt + (size_t)BH * NSORT;
  int*   inv = srt + (size_t)BH * NSORT;
  int*   off = inv + (size_t)BH * NSORT;                 // 32*257

  const int M = BSZ * LSEQ;                              // 16384

  embed_kernel<<<(M * DMODEL) / 256, 256, 0, stream>>>(
      x_enc, x_dec, x_mark_enc, x_mark_dec, token_conv, temporal_W, h);

  for (int lp = 0; lp < 2; ++lp) {
    // shared-QK and V projections
    gemm_wmma_kernel<<<dim3(DMODEL / 64, M / 128), 256, 0, stream>>>(
        h, LW(lp, 0), nullptr, tqk, M, DMODEL, DMODEL, 0);
    gemm_wmma_kernel<<<dim3(DMODEL / 64, M / 128), 256, 0, stream>>>(
        h, LW(lp, 1), nullptr, tv, M, DMODEL, DMODEL, 0);
    // LSH bucketing + stable counting sort
    hash_kernel<<<(BH * LSEQ) / 256, 256, 0, stream>>>(tqk, rotations, bkt);
    hist_kernel<<<BH, 256, 0, stream>>>(bkt, off);
    sort_kernel<<<BH * 256, 32, 0, stream>>>(bkt, off, srt, inv);
    // chunked attention with look-one-back
    lsh_attn_kernel<<<BH * NCH, 256, 0, stream>>>(tqk, tv, srt, big, slg);
    // unsort-free combine of hash rounds via inverse permutation
    combine_kernel<<<(BH * LSEQ) / 4, 256, 0, stream>>>(big, slg, inv, ta);
    // output projection + residual LN
    gemm_wmma_kernel<<<dim3(DMODEL / 64, M / 128), 256, 0, stream>>>(
        ta, LW(lp, 2), LW(lp, 3), tqk, M, DMODEL, DMODEL, 0);
    ln_kernel<<<M, 256, 0, stream>>>(h, tqk, LW(lp, 4), LW(lp, 5));
    // FFN: gelu(h@W1+b1)@W2+b2, residual LN
    gemm_wmma_kernel<<<dim3(DFF / 64, M / 128), 256, 0, stream>>>(
        h, LW(lp, 8), LW(lp, 9), big, M, DFF, DMODEL, 1);
    gemm_wmma_kernel<<<dim3(DMODEL / 64, M / 128), 256, 0, stream>>>(
        big, LW(lp, 10), LW(lp, 11), ta, M, DMODEL, DFF, 0);
    ln_kernel<<<M, 256, 0, stream>>>(h, ta, LW(lp, 6), LW(lp, 7));
  }

  ln_kernel<<<M, 256, 0, stream>>>(h, nullptr, ln_f_g, ln_f_b);
  proj_kernel<<<(BSZ * PREDL * 7 + 255) / 256, 256, 0, stream>>>(
      h, proj_W, proj_b, (float*)d_out);
}
